// Discriminator_41102837023008
// MI455X (gfx1250) — compile-verified, matrix operand order checked by reading, not run on hardware
//
#include <hip/hip_runtime.h>
#include <hip/hip_bf16.h>

// ---------------------------------------------------------------------------
// CDNA5 (gfx1250) WMMA types
// ---------------------------------------------------------------------------
typedef __attribute__((ext_vector_type(16))) _Float16 v16h;
typedef __attribute__((ext_vector_type(8)))  _Float16 v8h;
typedef __attribute__((ext_vector_type(8)))  float    v8f;
typedef int b128_t __attribute__((vector_size(16)));   // async-copy payload

#define GLOBAL_AS __attribute__((address_space(1)))
#define LDS_AS    __attribute__((address_space(3)))

#if __has_builtin(__builtin_amdgcn_global_load_async_to_lds_b128)
#define HAVE_ASYNC_LDS 1
#else
#define HAVE_ASYNC_LDS 0
#endif

__device__ __forceinline__ void wait_async_zero() {
#if HAVE_ASYNC_LDS
#if __has_builtin(__builtin_amdgcn_s_wait_asynccnt)
    __builtin_amdgcn_s_wait_asynccnt(0);
#else
    asm volatile("s_wait_asynccnt 0x0" ::: "memory");
#endif
#endif
}

__device__ __forceinline__ v8f wmma_f16(v16h a, v16h b, v8f c) {
    // v_wmma_f32_16x16x32_f16: D = A(16x32) * B(32x16) + C(16x16)
    return __builtin_amdgcn_wmma_f32_16x16x32_f16(
        /*neg_a=*/false, a, /*neg_b=*/false, b,
        /*c_mod=*/(short)0, c, /*reuse_a=*/false, /*reuse_b=*/false);
}

// A-fragment (16x32 f16) loader from a row-major [rows, ld] f16 matrix, tile
// at (r0, c0).  ISA layout: lanes 0-15 hold row (lane), K={c0..c0+7, c0+16..23};
// lanes 16-31 hold row (lane-16), K={c0+8..15, c0+24..31}.  Works equally for
// the B operand when the weight matrix is stored transposed ([N,K] row-major),
// since B^T has the same striping as A.
__device__ __forceinline__ v16h load_frag(const _Float16* __restrict__ base,
                                          int ld, int r0, int c0) {
    int l   = threadIdx.x & 31;
    int row = r0 + (l & 15);
    int cb  = c0 + ((l >> 4) << 3);           // +8 halves for upper lanes
    const _Float16* p = base + (size_t)row * ld + cb;
    v8h lo = *(const v8h*)(p);                // 16B contiguous
    v8h hi = *(const v8h*)(p + 16);           // next 16B group
    v16h out;
#pragma unroll
    for (int i = 0; i < 8; ++i) { out[i] = lo[i]; out[i + 8] = hi[i]; }
    return out;
}

// ---------------------------------------------------------------------------
// Generic fused WMMA GEMM:  C[M,N] = act(A[M,K] @ Bt[N,K]^T + bias)
// Block = 256 threads = 8 waves; block tile 64(M) x 128(N); each wave owns a
// 16x64 strip.  The 64x32 A tile is staged in LDS (double-buffered) via
// async global->LDS copies (ASYNCcnt) so the copy of tile k+1 overlaps the
// WMMAs on tile k and the two N-strips of waves share one A fetch.
// ---------------------------------------------------------------------------
template<int ACT, bool HAS_BIAS, bool OUT_F32, bool OUT_F16>
__global__ __launch_bounds__(256) void wmma_gemm_kernel(
        const _Float16* __restrict__ A, const _Float16* __restrict__ Bt,
        const float* __restrict__ bias, float* __restrict__ Cf,
        _Float16* __restrict__ Ch, int M, int N, int K)
{
    constexpr int LDA_S = 40;                      // padded stride: bank-safe
    __shared__ alignas(16) _Float16 sA[2][64 * LDA_S];

    int tid  = threadIdx.x;
    int wave = tid >> 5;
    int wm = wave & 3;                 // 4 waves along M
    int wn = wave >> 2;                // 2 waves along N
    int mblk = blockIdx.y * 64;
    int m0l  = wm * 16;                // local M offset inside A tile
    int n0   = blockIdx.x * 128 + wn * 64;

    // cooperative A-tile staging: thread t -> row t/4, 8 halves at (t%4)*8
    int lrow = tid >> 2;
    int lcol = (tid & 3) << 3;
    const _Float16* gA = A + (size_t)(mblk + lrow) * K + lcol;

    auto stage = [&](int buf, int k0) {
        _Float16* dst = &sA[buf][lrow * LDA_S + lcol];
#if HAVE_ASYNC_LDS
        __builtin_amdgcn_global_load_async_to_lds_b128(
            (GLOBAL_AS b128_t*)(gA + k0), (LDS_AS b128_t*)dst, 0, 0);
#else
        *(v8h*)dst = *(const v8h*)(gA + k0);
#endif
    };

    stage(0, 0);

    int l = tid & 31;
    v8f acc[4] = {};
    int buf = 0;
    for (int k0 = 0; k0 < K; k0 += 32) {
        wait_async_zero();             // my async copy done (per-wave counter)
        __syncthreads();               // all waves' copies visible
        if (k0 + 32 < K) {
            stage(buf ^ 1, k0 + 32);   // overlap next copy with this compute
            __builtin_prefetch(Bt + (size_t)n0 * K + k0 + 32, 0, 0);
        }

        // A fragment from LDS tile
        const _Float16* pa =
            &sA[buf][(m0l + (l & 15)) * LDA_S + ((l >> 4) << 3)];
        v8h lo = *(const v8h*)pa;
        v8h hi = *(const v8h*)(pa + 16);
        v16h a;
#pragma unroll
        for (int i = 0; i < 8; ++i) { a[i] = lo[i]; a[i + 8] = hi[i]; }

#pragma unroll
        for (int j = 0; j < 4; ++j) {
            v16h b = load_frag(Bt, K, n0 + 16 * j, k0);   // weights: L2-hot
            acc[j] = wmma_f16(a, b, acc[j]);
        }
        buf ^= 1;
    }

    // Epilogue: C/D layout -> lane l, vgpr v : row = m0 + v + (l>=16)*8,
    // col = n0 + 16*j + (l&15)
    int rr  = mblk + m0l + ((l >> 4) << 3);
    int cc0 = l & 15;
#pragma unroll
    for (int j = 0; j < 4; ++j) {
        int cc = n0 + 16 * j + cc0;
        float bv = 0.0f;
        if (HAS_BIAS) bv = bias[cc];
#pragma unroll
        for (int v = 0; v < 8; ++v) {
            float x = acc[j][v] + bv;
            if (ACT == 1) x = (x >= 0.0f) ? x : 0.01f * x;   // LeakyReLU
            if (ACT == 2) x = (x > 0.0f) ? x : 0.0f;         // ReLU
            size_t idx = (size_t)(rr + v) * N + cc;
            if (OUT_F32) Cf[idx] = x;
            if (OUT_F16) Ch[idx] = (_Float16)x;
        }
    }
}

// ---------------------------------------------------------------------------
// vec[t,j] = emb[marker[t]][j] + time[t]*w_time[j] + b_time[j]   (f16 out)
// ---------------------------------------------------------------------------
__global__ __launch_bounds__(256) void build_vec_kernel(
        const int* __restrict__ marker, const float* __restrict__ time,
        const float* __restrict__ emb, const float* __restrict__ w_time,
        const float* __restrict__ b_time, _Float16* __restrict__ out, int Ntok)
{
    int idx = blockIdx.x * 256 + threadIdx.x;
    if (idx >= Ntok * 256) return;
    int t = idx >> 8, j = idx & 255;
    int m = marker[t];
    float v = emb[(size_t)m * 256 + j] + time[t] * w_time[j] + b_time[j];
    out[idx] = (_Float16)v;
}

// Weight f32 [K,N] -> f16 transposed [N,K]
__global__ __launch_bounds__(256) void transpose_w_kernel(
        const float* __restrict__ W, _Float16* __restrict__ Wt, int K, int N)
{
    int idx = blockIdx.x * 256 + threadIdx.x;
    if (idx >= K * N) return;
    int n = idx / K, k = idx - n * K;
    Wt[idx] = (_Float16)W[(size_t)k * N + n];
}

// V [nseq*512, 256] f16 -> Vt [nseq][4][64][512] f16 (per-head transpose)
__global__ __launch_bounds__(256) void transpose_v_kernel(
        const _Float16* __restrict__ Vf, _Float16* __restrict__ Vt, int nseq)
{
    int idx = blockIdx.x * 256 + threadIdx.x;
    if (idx >= nseq * 512 * 256) return;
    int tok = idx >> 8, c = idx & 255;
    int n = tok >> 9, t = tok & 511;
    int h = c >> 6, d = c & 63;
    Vt[((size_t)((n * 4 + h) * 64 + d) << 9) + t] = Vf[idx];
}

// ---------------------------------------------------------------------------
// Causal attention, one wave per (seq, head, 16-row q tile).
// scores & softmax in LDS; QK^T and P@V via WMMA.
// ---------------------------------------------------------------------------
__global__ __launch_bounds__(32) void attn_kernel(
        const _Float16* __restrict__ Q, const _Float16* __restrict__ Km,
        const _Float16* __restrict__ Vt, _Float16* __restrict__ O)
{
    __shared__ alignas(16) float    sS[16 * 512];   // scores (f32)
    __shared__ alignas(16) _Float16 sP[16 * 512];   // softmax probs (f16)

    int qt = blockIdx.x;            // q tile (0..31)
    int h  = blockIdx.y;            // head
    int n  = blockIdx.z;            // sequence
    int qr = qt * 16;
    int l  = threadIdx.x;           // lane 0..31

    const _Float16* Qb = Q  + (size_t)n * 512 * 256;
    const _Float16* Kb = Km + (size_t)n * 512 * 256;
    const _Float16* Vb = Vt + ((size_t)(n * 4 + h) * 64) * 512;

    // init scores to -inf (uncomputed causal region -> exp() == 0)
    for (int i = l; i < 16 * 512; i += 32) sS[i] = -3.0e38f;

    // Q fragments for this 16-row tile (K-dim = 64 -> two 32-chunks)
    v16h a0 = load_frag(Qb, 256, qr, h * 64 + 0);
    v16h a1 = load_frag(Qb, 256, qr, h * 64 + 32);

    int rr  = (l >> 4) << 3;        // local-row base for C layout
    int cc0 = l & 15;

    int kmax = qr + 16;             // causal: keys 0 .. qr+15
    for (int k0 = 0; k0 < kmax; k0 += 16) {
        v8f acc = {};
        v16h b0 = load_frag(Kb, 256, k0, h * 64 + 0);
        v16h b1 = load_frag(Kb, 256, k0, h * 64 + 32);
        acc = wmma_f16(a0, b0, acc);
        acc = wmma_f16(a1, b1, acc);
        int cc = k0 + cc0;
#pragma unroll
        for (int v = 0; v < 8; ++v) {
            int row = rr + v;
            float sc = acc[v] * 0.125f;                 // 1/sqrt(64)
            sS[row * 512 + cc] = (cc <= qr + row) ? sc : -3.0e38f;
        }
    }

    // row softmax (single wave -> DS ops in order, no barrier needed)
    for (int r = 0; r < 16; ++r) {
        float mx = -3.0e38f;
        for (int c = l; c < 512; c += 32) mx = fmaxf(mx, sS[r * 512 + c]);
        for (int o = 16; o > 0; o >>= 1) mx = fmaxf(mx, __shfl_xor(mx, o, 32));
        float sum = 0.0f;
        for (int c = l; c < 512; c += 32) {
            float e = __expf(sS[r * 512 + c] - mx);
            sS[r * 512 + c] = e;
            sum += e;
        }
        for (int o = 16; o > 0; o >>= 1) sum += __shfl_xor(sum, o, 32);
        float inv = 1.0f / sum;
        for (int c = l; c < 512; c += 32)
            sP[r * 512 + c] = (_Float16)(sS[r * 512 + c] * inv);
    }

    // O(16x64) = P(16x512) @ V(512x64), Vt gives contiguous B^T fragments
    v8f oacc[4] = {};
    int kend = (kmax + 31) & ~31;
    for (int k0 = 0; k0 < kend; k0 += 32) {
        v16h pa = load_frag(sP, 512, 0, k0);
#pragma unroll
        for (int j = 0; j < 4; ++j) {
            v16h vb = load_frag(Vb, 512, j * 16, k0);
            oacc[j] = wmma_f16(pa, vb, oacc[j]);
        }
    }
#pragma unroll
    for (int j = 0; j < 4; ++j)
#pragma unroll
        for (int v = 0; v < 8; ++v)
            O[(size_t)(n * 512 + qr + rr + v) * 256 + h * 64 + j * 16 + cc0] =
                (_Float16)oacc[j][v];
}

// ---------------------------------------------------------------------------
// LayerNorm over rows of 256: out = LN(X + R) * g + b  (f32 + optional f16)
// 8 rows per block, one wave per row.
// ---------------------------------------------------------------------------
__global__ __launch_bounds__(256) void ln_kernel(
        const float* __restrict__ X, const float* __restrict__ R,
        const float* __restrict__ g, const float* __restrict__ b,
        float* __restrict__ outF, _Float16* __restrict__ outH, int Ntok)
{
    int wave = threadIdx.x >> 5, l = threadIdx.x & 31;
    int row = blockIdx.x * 8 + wave;
    if (row >= Ntok) return;
    const float* xr = X + (size_t)row * 256;
    const float* rr = R + (size_t)row * 256;
    float vals[8];
    float s = 0.0f;
#pragma unroll
    for (int i = 0; i < 8; ++i) {
        int c = l + 32 * i;
        vals[i] = xr[c] + rr[c];
        s += vals[i];
    }
    for (int o = 16; o > 0; o >>= 1) s += __shfl_xor(s, o, 32);
    float mean = s * (1.0f / 256.0f);
    float vs = 0.0f;
#pragma unroll
    for (int i = 0; i < 8; ++i) { float d = vals[i] - mean; vs += d * d; }
    for (int o = 16; o > 0; o >>= 1) vs += __shfl_xor(vs, o, 32);
    float inv = rsqrtf(vs * (1.0f / 256.0f) + 1e-5f);
#pragma unroll
    for (int i = 0; i < 8; ++i) {
        int c = l + 32 * i;
        float y = (vals[i] - mean) * inv * g[c] + b[c];
        outF[(size_t)row * 256 + c] = y;
        if (outH) outH[(size_t)row * 256 + c] = (_Float16)y;
    }
}

// reward = sigmoid(h2 . w_out + b_out) * mask
__global__ __launch_bounds__(256) void out_kernel(
        const float* __restrict__ H2, const float* __restrict__ w_out,
        const float* __restrict__ b_out, const float* __restrict__ mask,
        float* __restrict__ out, int Ntok)
{
    int wave = threadIdx.x >> 5, l = threadIdx.x & 31;
    int row = blockIdx.x * 8 + wave;
    if (row >= Ntok) return;
    const float* h = H2 + (size_t)row * 256;
    float s = 0.0f;
#pragma unroll
    for (int i = 0; i < 8; ++i) s += h[l + 32 * i] * w_out[l + 32 * i];
    for (int o = 16; o > 0; o >>= 1) s += __shfl_xor(s, o, 32);
    if (l == 0) {
        float r = 1.0f / (1.0f + __expf(-(s + b_out[0])));
        out[row] = r * mask[row];
    }
}

__global__ __launch_bounds__(256) void copy_kernel(
        const float* __restrict__ src, float* __restrict__ dst, int n)
{
    int i = blockIdx.x * 256 + threadIdx.x;
    if (i < n) dst[i] = src[i];
}

// ---------------------------------------------------------------------------
// Host driver
// ---------------------------------------------------------------------------
extern "C" void kernel_launch(void* const* d_in, const int* in_sizes, int n_in,
                              void* d_out, int out_size, void* d_ws, size_t ws_size,
                              hipStream_t stream)
{
    (void)in_sizes; (void)n_in; (void)out_size; (void)ws_size;

    const int*   real_marker = (const int*)  d_in[0];
    const float* real_time   = (const float*)d_in[1];
    const float* real_mask   = (const float*)d_in[2];
    const int*   fake_marker = (const int*)  d_in[3];
    const float* fake_time   = (const float*)d_in[4];
    const float* fake_mask   = (const float*)d_in[5];
    const float* emb         = (const float*)d_in[6];
    const float* w_time      = (const float*)d_in[7];
    const float* b_time      = (const float*)d_in[8];
    const float* w_embed     = (const float*)d_in[9];
    const float* b_embed     = (const float*)d_in[10];
    const float* wq          = (const float*)d_in[11];
    const float* wk          = (const float*)d_in[12];
    const float* wv          = (const float*)d_in[13];
    const float* w_ao        = (const float*)d_in[14];
    const float* b_ao        = (const float*)d_in[15];
    const float* ln1_g       = (const float*)d_in[16];
    const float* ln1_b       = (const float*)d_in[17];
    const float* w_ff1       = (const float*)d_in[18];
    const float* b_ff1       = (const float*)d_in[19];
    const float* w_ff2       = (const float*)d_in[20];
    const float* b_ff2       = (const float*)d_in[21];
    const float* ln2_g       = (const float*)d_in[22];
    const float* ln2_b       = (const float*)d_in[23];
    const float* w_out       = (const float*)d_in[24];
    const float* b_out       = (const float*)d_in[25];
    float* out = (float*)d_out;

    const int B = 64, T = 512, D = 256, DFF = 1024;
    const int NSEQ = B;                 // per pass
    const int NTOK = NSEQ * T;          // 32768 per pass

    // --- workspace layout (bytes), aliased across pipeline phases ----------
    char* ws = (char*)d_ws;
    size_t off = 0;
    auto alloc = [&](size_t bytes) { char* p = ws + off; off += (bytes + 255) & ~(size_t)255; return p; };

    _Float16* WEMB_T = (_Float16*)alloc((size_t)D * D * 2);
    _Float16* WQ_T   = (_Float16*)alloc((size_t)D * D * 2);
    _Float16* WK_T   = (_Float16*)alloc((size_t)D * D * 2);
    _Float16* WV_T   = (_Float16*)alloc((size_t)D * D * 2);
    _Float16* WAO_T  = (_Float16*)alloc((size_t)D * D * 2);
    _Float16* WFF1_T = (_Float16*)alloc((size_t)D * DFF * 2);
    _Float16* WFF2_T = (_Float16*)alloc((size_t)D * DFF * 2);

    _Float16* VEC16  = (_Float16*)alloc((size_t)NTOK * D * 2);   // alias: O f16
    float*    XF32   = (float*)   alloc((size_t)NTOK * D * 4);   // alias: F2 f32
    _Float16* XF16   = (_Float16*)alloc((size_t)NTOK * D * 2);   // alias: Vt
    _Float16* QF16   = (_Float16*)alloc((size_t)NTOK * D * 2);   // alias: H f16
    _Float16* KF16   = (_Float16*)alloc((size_t)NTOK * D * 2);
    _Float16* VF16   = (_Float16*)alloc((size_t)NTOK * D * 2);
    float*    AOF32  = (float*)   alloc((size_t)NTOK * D * 4);   // alias: H2 f32
    float*    HF32   = (float*)   alloc((size_t)NTOK * D * 4);
    _Float16* F1F16  = (_Float16*)alloc((size_t)NTOK * DFF * 2);
    _Float16* OF16   = VEC16;
    _Float16* VT16   = XF16;
    _Float16* HF16   = QF16;
    float*    F2F32  = XF32;
    float*    H2F32  = AOF32;

    // --- one-time weight convert/transpose (f32 [K,N] -> f16 [N,K]) --------
    {
        int n1 = D * D, n2 = D * DFF;
        transpose_w_kernel<<<(n1 + 255) / 256, 256, 0, stream>>>(w_embed, WEMB_T, D, D);
        transpose_w_kernel<<<(n1 + 255) / 256, 256, 0, stream>>>(wq,      WQ_T,   D, D);
        transpose_w_kernel<<<(n1 + 255) / 256, 256, 0, stream>>>(wk,      WK_T,   D, D);
        transpose_w_kernel<<<(n1 + 255) / 256, 256, 0, stream>>>(wv,      WV_T,   D, D);
        transpose_w_kernel<<<(n1 + 255) / 256, 256, 0, stream>>>(w_ao,    WAO_T,  D, D);
        transpose_w_kernel<<<(n2 + 255) / 256, 256, 0, stream>>>(w_ff1,   WFF1_T, D, DFF);
        transpose_w_kernel<<<(n2 + 255) / 256, 256, 0, stream>>>(w_ff2,   WFF2_T, DFF, D);
    }

    // --- 3 passes: real, fake[0], fake[1] ----------------------------------
    for (int p = 0; p < 3; ++p) {
        const int*   marker; const float* tim; const float* msk; float* rout;
        if (p == 0) { marker = real_marker;            tim = real_time;            msk = real_mask;            rout = out; }
        else {
            int o = (p - 1) * NTOK;
            marker = fake_marker + o; tim = fake_time + o; msk = fake_mask + o;
            rout = out + 2 * NTOK + (p - 1) * NTOK;   // after real_rewards+real_mask
        }

        build_vec_kernel<<<(NTOK * D) / 256, 256, 0, stream>>>(
            marker, tim, emb, w_time, b_time, VEC16, NTOK);

        dim3 gD(D / 128, NTOK / 64), gF(DFF / 128, NTOK / 64);
        // x = LeakyReLU(vec @ w_embed + b_embed)   -> f32 + f16
        wmma_gemm_kernel<1, true, true, true><<<gD, 256, 0, stream>>>(
            VEC16, WEMB_T, b_embed, XF32, XF16, NTOK, D, D);
        // q,k,v = x @ W  (no bias)
        wmma_gemm_kernel<0, false, false, true><<<gD, 256, 0, stream>>>(
            XF16, WQ_T, nullptr, nullptr, QF16, NTOK, D, D);
        wmma_gemm_kernel<0, false, false, true><<<gD, 256, 0, stream>>>(
            XF16, WK_T, nullptr, nullptr, KF16, NTOK, D, D);
        wmma_gemm_kernel<0, false, false, true><<<gD, 256, 0, stream>>>(
            XF16, WV_T, nullptr, nullptr, VF16, NTOK, D, D);
        // per-head V transpose for contiguous B-fragments in P@V
        transpose_v_kernel<<<(NTOK * D) / 256, 256, 0, stream>>>(VF16, VT16, NSEQ);
        // causal attention
        attn_kernel<<<dim3(T / 16, 4, NSEQ), 32, 0, stream>>>(QF16, KF16, VT16, OF16);
        // o @ w_ao + b_ao
        wmma_gemm_kernel<0, true, true, false><<<gD, 256, 0, stream>>>(
            OF16, WAO_T, b_ao, AOF32, nullptr, NTOK, D, D);
        // h = LN(x + ao)
        ln_kernel<<<NTOK / 8, 256, 0, stream>>>(AOF32, XF32, ln1_g, ln1_b, HF32, HF16, NTOK);
        // ff = relu(h @ w_ff1 + b_ff1)
        wmma_gemm_kernel<2, true, false, true><<<gF, 256, 0, stream>>>(
            HF16, WFF1_T, b_ff1, nullptr, F1F16, NTOK, DFF, D);
        // f2 = ff @ w_ff2 + b_ff2
        wmma_gemm_kernel<0, true, true, false><<<gD, 256, 0, stream>>>(
            F1F16, WFF2_T, b_ff2, F2F32, nullptr, NTOK, D, DFF);
        // h2 = LN(h + f2)
        ln_kernel<<<NTOK / 8, 256, 0, stream>>>(F2F32, HF32, ln2_g, ln2_b, H2F32, nullptr, NTOK);
        // reward = sigmoid(h2 @ w_out + b_out) * mask
        out_kernel<<<NTOK / 8, 256, 0, stream>>>(H2F32, w_out, b_out, msk, rout, NTOK);
    }

    // output tuple: [real_rewards | real_mask | fake_rewards | fake_mask]
    copy_kernel<<<(NTOK + 255) / 256, 256, 0, stream>>>(real_mask, out + NTOK, NTOK);
    copy_kernel<<<(2 * NTOK + 255) / 256, 256, 0, stream>>>(fake_mask, out + 4 * NTOK, 2 * NTOK);
}
